// make_visual_language_merger_edge_57655640982179
// MI455X (gfx1250) — matrix operand order repaired and checked by reading, not compiled, not run on hardware
//
#include <hip/hip_runtime.h>
#include <hip/hip_bf16.h>
#include <stddef.h>

typedef float     v8f  __attribute__((ext_vector_type(8)));
typedef float     v4f  __attribute__((ext_vector_type(4)));
typedef _Float16  v16h __attribute__((ext_vector_type(16)));
typedef _Float16  v8h  __attribute__((ext_vector_type(8)));

#define CCH   256      // channels
#define EDIM  200      // embedding dim
#define KDIM  2304     // 256 * 9
#define BNEPS 1e-5f

// ---------------------------------------------------------------- K0: W2 -> f16
// conv2_w is [Cout=256][Cin=256][3][3] f32; flattened it is exactly
// Bt[n=Cout][k=Cin*9+uv] row-major, which is the layout the WMMA B-fragment wants.
__global__ __launch_bounds__(256) void k_cast_w2(const float* __restrict__ w2,
                                                 _Float16* __restrict__ w2h, int n) {
    int i = blockIdx.x * 256 + threadIdx.x;
    if (i < n) w2h[i] = (_Float16)w2[i];
}

// ---------------------------------------------------------------- K1: pair features
// One block (256 threads) per relation pair. Threads 0..9 build the separable
// AvgPool window sums (outer-product trick); thread c computes conv1+BN1+ReLU
// for channel c at the 3x3 positions conv2 actually reads; the 2304-half row is
// staged in LDS and written back with coalesced dword stores.
__global__ __launch_bounds__(256) void k_pair_feat(
    const float* __restrict__ sub_emb, const float* __restrict__ obj_emb,
    const float* __restrict__ w1, const float* __restrict__ b1,
    const float* __restrict__ g1, const float* __restrict__ be1,
    const float* __restrict__ m1, const float* __restrict__ v1,
    const int*   __restrict__ obj_labels, const int* __restrict__ pairs,
    _Float16* __restrict__ Afeat)
{
    __shared__ float ps[5];            // row window sums (scaled by 1/841)
    __shared__ float po[5];            // col window sums
    __shared__ _Float16 rowbuf[KDIM];  // staged A row (4.5 KB)
    const int r = blockIdx.x;
    const int t = threadIdx.x;

    if (t < 10) {
        const int i  = (t < 5) ? t : (t - 5);
        const int n0 = pairs[2 * r + ((t < 5) ? 0 : 1)];
        const float* row = ((t < 5) ? sub_emb : obj_emb) + (size_t)obj_labels[n0] * EDIM;
        int lo = 29 * i - 2; if (lo < 0) lo = 0;   // padding=2, zeros contribute 0
        const int hi = 29 * i + 26;                // i<=4 -> hi<=142 < 200
        float s = 0.f;
        for (int k = lo; k <= hi; ++k) s += row[k];
        if (t < 5) ps[i] = s * (1.f / 841.f); else po[i] = s;
    }
    __syncthreads();

    const int c = t;
    float w[9];
#pragma unroll
    for (int k = 0; k < 9; ++k) w[k] = w1[c * 9 + k];
    const float inv  = g1[c] * rsqrtf(v1[c] + BNEPS);
    const float base = be1[c] + (b1[c] - m1[c]) * inv;

    // tav[a][v] = sum_b w[a,b] * po[v+b]
    float tav[3][3];
#pragma unroll
    for (int a = 0; a < 3; ++a)
#pragma unroll
        for (int v = 0; v < 3; ++v)
            tav[a][v] = w[a * 3 + 0] * po[v] + w[a * 3 + 1] * po[v + 1] + w[a * 3 + 2] * po[v + 2];

#pragma unroll
    for (int u = 0; u < 3; ++u)
#pragma unroll
        for (int v = 0; v < 3; ++v) {
            float x = ps[u] * tav[0][v] + ps[u + 1] * tav[1][v] + ps[u + 2] * tav[2][v];
            float y = fmaxf(x * inv + base, 0.f);
            rowbuf[c * 9 + u * 3 + v] = (_Float16)y;
        }
    __syncthreads();

    // coalesced writeback: 1152 dwords, 256 threads
    const unsigned* src = (const unsigned*)rowbuf;
    unsigned* dst = (unsigned*)(Afeat + (size_t)r * KDIM);
    for (int i = t; i < KDIM / 2; i += 256) dst[i] = src[i];
}

// ---------------------------------------------------------------- K2: WMMA GEMM + BN2 + ReLU
// M = R pairs, N = 256 channels, K = 2304.  16 waves/block; wave w owns N-tile w,
// block owns one M-tile of 16 (A fragments shared via WGP$).
__global__ __launch_bounds__(512) void k_gemm_att(
    const _Float16* __restrict__ A,    // [R, 2304] row-major
    const _Float16* __restrict__ Bt,   // [256, 2304] row-major (= conv2_w in f16)
    const float* __restrict__ g2, const float* __restrict__ be2,
    const float* __restrict__ m2, const float* __restrict__ v2,
    const float* __restrict__ b2,
    float* __restrict__ att)           // [R, 256]
{
    const int lane  = threadIdx.x & 31;
    const int wave  = threadIdx.x >> 5;          // 0..15 -> N-tile
    const int mbase = blockIdx.x * 16;
    const int half  = lane >> 4;                 // lane group 0/1
    const int l15   = lane & 15;

    // CDNA5 f16 A-fragment: lanes0-15 K = {0..7, 16..23}; lanes16-31 K = +8
    const _Float16* Arow = A  + (size_t)(mbase + l15) * KDIM + half * 8;
    // CDNA5 f16 B-fragment: N = lane&15, K = {0..7, 16..23} + 8*half (same as A)
    const _Float16* Brow = Bt + (size_t)(wave * 16 + l15) * KDIM + half * 8;

    v8f acc = {0.f, 0.f, 0.f, 0.f, 0.f, 0.f, 0.f, 0.f};
    for (int k = 0; k < KDIM; k += 32) {
        union { v16h v; v8h h[2]; } a;
        a.h[0] = *(const v8h*)(Arow + k);
        a.h[1] = *(const v8h*)(Arow + k + 16);
        union { v16h v; v8h h[2]; } bu;
        bu.h[0] = *(const v8h*)(Brow + k);
        bu.h[1] = *(const v8h*)(Brow + k + 16);
        v16h b = bu.v;
        __builtin_prefetch(Arow + k + 64, 0, 1);  // global_prefetch_b8
        __builtin_prefetch(Brow + k + 64, 0, 1);
        acc = __builtin_amdgcn_wmma_f32_16x16x32_f16(
            /*neg_a=*/false, a.v, /*neg_b=*/false, b,
            /*c_mod=*/(short)0, acc, /*reuse_a=*/false, /*reuse_b=*/false);
    }

    // fused BN2 + conv2 bias + ReLU; channel depends only on lane
    const int   ch   = wave * 16 + l15;          // N = lane&15
    const float inv  = g2[ch] * rsqrtf(v2[ch] + BNEPS);
    const float base = be2[ch] + (b2[ch] - m2[ch]) * inv;

    float* drow = att + (size_t)mbase * CCH + ch;
#pragma unroll
    for (int j = 0; j < 8; ++j) {                // C/D layout: M = j + 8*half
        const int M = j + half * 8;
        drow[(size_t)M * CCH] = fmaxf(acc[j] * inv + base, 0.f);
    }
}

// ---------------------------------------------------------------- K3: streaming fuse
// out[r,c,h,w] = visual_feature[r,c,h,w] * att[r,c]  — the HBM roofline stage.
// float4-wide, non-temporal in/out (each byte touched exactly once; keep L2 for att).
// A float4 straddles at most one 49-float (r,c)-plane boundary.
__global__ __launch_bounds__(256) void k_fuse(const v4f* __restrict__ vf,
                                              const float* __restrict__ att,
                                              float* __restrict__ out, int total4) {
    int i = blockIdx.x * 256 + threadIdx.x;
    if (i >= total4) return;
    const int base = i * 4;
    const int rc0  = base / 49;                  // mul-shift
    const int rem  = base - rc0 * 49;            // 0..48
    const float a0 = __ldg(att + rc0);
    const float a1 = (rem >= 46) ? __ldg(att + rc0 + 1) : a0;

    v4f v = __builtin_nontemporal_load(vf + i);
    v4f o;
    o.x = v.x * a0;
    o.y = v.y * ((rem >= 48) ? a1 : a0);
    o.z = v.z * ((rem >= 47) ? a1 : a0);
    o.w = v.w * ((rem >= 46) ? a1 : a0);
    __builtin_nontemporal_store(o, (v4f*)(out) + i);
}

// ---------------------------------------------------------------- launcher
extern "C" void kernel_launch(void* const* d_in, const int* in_sizes, int n_in,
                              void* d_out, int out_size, void* d_ws, size_t ws_size,
                              hipStream_t stream) {
    const float* visual   = (const float*)d_in[0];
    const float* sub_emb  = (const float*)d_in[1];
    const float* obj_emb  = (const float*)d_in[2];
    const float* conv1_w  = (const float*)d_in[3];
    const float* conv1_b  = (const float*)d_in[4];
    const float* bn1_g    = (const float*)d_in[5];
    const float* bn1_b    = (const float*)d_in[6];
    const float* bn1_m    = (const float*)d_in[7];
    const float* bn1_v    = (const float*)d_in[8];
    const float* conv2_w  = (const float*)d_in[9];
    const float* conv2_b  = (const float*)d_in[10];
    const float* bn2_g    = (const float*)d_in[11];
    const float* bn2_b    = (const float*)d_in[12];
    const float* bn2_m    = (const float*)d_in[13];
    const float* bn2_v    = (const float*)d_in[14];
    const int*   labels   = (const int*)d_in[15];
    const int*   pairs    = (const int*)d_in[16];
    float*       out      = (float*)d_out;

    const int R   = in_sizes[16] / 2;             // 4096
    const int W2N = in_sizes[9];                  // 256*256*9 = 589824

    // workspace layout (offsets 32B-aligned): W2 f16 | A f16 | att f32  (~24.3 MB)
    char* ws = (char*)d_ws;
    const size_t offA   = (size_t)W2N * sizeof(_Float16);             // 1,179,648
    const size_t offAtt = offA + (size_t)R * KDIM * sizeof(_Float16); // +18,874,368
    _Float16* W2h   = (_Float16*)ws;
    _Float16* Afeat = (_Float16*)(ws + offA);
    float*    attB  = (float*)(ws + offAtt);

    k_cast_w2<<<(W2N + 255) / 256, 256, 0, stream>>>(conv2_w, W2h, W2N);

    k_pair_feat<<<R, 256, 0, stream>>>(sub_emb, obj_emb, conv1_w, conv1_b,
                                       bn1_g, bn1_b, bn1_m, bn1_v,
                                       labels, pairs, Afeat);

    k_gemm_att<<<R / 16, 512, 0, stream>>>(Afeat, W2h,
                                           bn2_g, bn2_b, bn2_m, bn2_v, conv2_b,
                                           attB);

    const int total4 = (R * CCH * 49) / 4;        // divisible: R*CCH % 4 == 0
    k_fuse<<<(total4 + 255) / 256, 256, 0, stream>>>((const v4f*)visual, attB, out, total4);
}
